// Nearest_36575941493245
// MI455X (gfx1250) — matrix operand order repaired and checked by reading, not compiled, not run on hardware
//
#include <hip/hip_runtime.h>

typedef __attribute__((ext_vector_type(16))) __bf16 v16bf;
typedef __attribute__((ext_vector_type(8)))  float  v8f;

#define B_  64
#define K_  64
#define N_  16
#define M_  512
#define D_  512

// Workspace layout (bytes):
//   [0,          33554432)   test_bf16   : B*M*D   bf16   (33.5 MB)
//   [33554432,  100663296)   train_bf16  : B*K*N*D bf16   (67 MB)
//   [100663296, 100925440)   train_sq    : B*K*N   f32
//   [100925440, 101056512)   test_sq     : B*M     f32
#define WS_TESTBF   0
#define WS_TRAINBF  33554432
#define WS_TRAINSQ  100663296
#define WS_TESTSQ   100925440

static __device__ __forceinline__ unsigned int f2bf1(float f) {
  unsigned int u = __builtin_bit_cast(unsigned int, f);
  return (u + 0x7FFFu + ((u >> 16) & 1u)) >> 16;   // round-to-nearest-even
}

// Generic->LDS offset: flat LDS aperture maps addr[31:0] to the wave-relative
// LDS byte offset, so truncation gives the DS/async address.
static __device__ __forceinline__ unsigned lds_off(const void* p) {
  return (unsigned)(unsigned long long)p;
}

// CDNA5 async copy: global -> LDS, 16B per lane, tracked by ASYNCcnt.
// GVS addressing: saddr = uniform 64-bit base, vaddr = per-lane byte offset.
static __device__ __forceinline__ void async_copy_b128(
    unsigned lds_byte, const unsigned short* base, unsigned byte_off) {
  asm volatile("global_load_async_to_lds_b128 %0, %1, %2"
               :: "v"(lds_byte), "v"(byte_off), "s"(base)
               : "memory");
}

// ---------------------------------------------------------------------------
// Prep: fp32 -> bf16 conversion (done ONCE, not per-use) + fused row sum-sq.
// One wave per row of 512 floats. 98304 rows (32768 test + 65536 train).
// ---------------------------------------------------------------------------
__global__ __launch_bounds__(256) void prep_kernel(
    const float* __restrict__ train, const float* __restrict__ test,
    unsigned short* __restrict__ testbf, unsigned short* __restrict__ trainbf,
    float* __restrict__ trainsq, float* __restrict__ testsq) {
  const int tid  = threadIdx.x;
  const int lane = tid & 31;
  const int row  = (blockIdx.x * 256 + tid) >> 5;  // global wave id

  const float*    src;
  unsigned short* dstb;
  float*          dsq;
  if (row < B_ * M_) {                 // test rows
    src  = test   + (size_t)row * D_;
    dstb = testbf + (size_t)row * D_;
    dsq  = testsq + row;
  } else {                             // train rows
    const int r = row - B_ * M_;
    src  = train   + (size_t)r * D_;
    dstb = trainbf + (size_t)r * D_;
    dsq  = trainsq + r;
  }

  float s = 0.0f;
  const float4* s4 = (const float4*)src;
  uint2*        d2 = (uint2*)dstb;
#pragma unroll
  for (int j = 0; j < 4; ++j) {
    const int i4 = lane + j * 32;      // coalesced float4 loads
    float4 x = s4[i4];
    s += x.x * x.x + x.y * x.y + x.z * x.z + x.w * x.w;
    uint2 p;
    p.x = f2bf1(x.x) | (f2bf1(x.y) << 16);
    p.y = f2bf1(x.z) | (f2bf1(x.w) << 16);
    d2[i4] = p;                        // coalesced 8B bf16 stores
  }
#pragma unroll
  for (int m = 16; m >= 1; m >>= 1) s += __shfl_xor(s, m);
  if (lane == 0) *dsq = s;
}

// ---------------------------------------------------------------------------
// Main: bf16 WMMA GEMM + fused (2*dot - train_sq - test_sq) + max-over-shots.
// WG = 256 thr (8 waves). WG tile: 128 test rows x 4 classes (64 shot-cols).
// Wave tile: 32 rows x 32 cols = 2x2 WMMA accumulators.
// D processed in chunks of 64 with double-buffered LDS + async global->LDS.
// ---------------------------------------------------------------------------
__global__ __launch_bounds__(256) void nearest_wmma_kernel(
    const unsigned short* __restrict__ testbf,
    const unsigned short* __restrict__ trainbf,
    const float* __restrict__ trainsq,
    const float* __restrict__ testsq,
    float* __restrict__ out) {
  // rows: 64 bf16 chunk = 128B data, padded to 144B (9 uint4) vs bank conflicts
  __shared__ uint4 ldsA[2][128 * 9];   // 2 x 18KB (test tiles)
  __shared__ uint4 ldsB[2][64 * 9];    // 2 x  9KB (train tiles)

  const int tid  = threadIdx.x;
  const int lane = tid & 31;
  const int wave = tid >> 5;
  const int wm   = wave & 3;        // 4 m-subblocks of 32 rows
  const int wc   = wave >> 2;       // 2 class-pair subblocks

  const int m0     = blockIdx.x * 128;  // 0..3
  const int class0 = blockIdx.y * 4;    // 0..15
  const int b      = blockIdx.z;        // batch

  const unsigned short* testBase  = testbf  + ((size_t)b * M_ + m0) * D_;
  const unsigned short* trainBase = trainbf + ((size_t)b * (K_ * N_) + class0 * N_) * D_;

  const unsigned ldsAoff0 = lds_off(&ldsA[0][0]);
  const unsigned ldsAoff1 = lds_off(&ldsA[1][0]);
  const unsigned ldsBoff0 = lds_off(&ldsB[0][0]);
  const unsigned ldsBoff1 = lds_off(&ldsB[1][0]);

  v8f zero = {};
  v8f acc[2][2];
#pragma unroll
  for (int i = 0; i < 2; ++i)
#pragma unroll
    for (int j = 0; j < 2; ++j) acc[i][j] = zero;

  const int hi  = lane >> 4;   // 0 | 1 : lane-half select
  const int l15 = lane & 15;

  // ---- async stage of one 64-wide D chunk: 6 b128 copies per thread ----
  auto stage_tiles = [&](unsigned aOff, unsigned bOff, int d0) {
#pragma unroll
    for (int j = 0; j < 4; ++j) {          // A: 1024 slots / 256 thr
      const int s = tid + j * 256;
      const int r = s >> 3, q = s & 7;
      async_copy_b128(aOff + (unsigned)(r * 9 + q) * 16u, testBase,
                      (unsigned)((r * D_ + d0 + q * 8) * 2));
    }
#pragma unroll
    for (int j = 0; j < 2; ++j) {          // B: 512 slots / 256 thr
      const int s = tid + j * 256;
      const int r = s >> 3, q = s & 7;
      async_copy_b128(bOff + (unsigned)(r * 9 + q) * 16u, trainBase,
                      (unsigned)((r * D_ + d0 + q * 8) * 2));
    }
  };

  stage_tiles(ldsAoff0, ldsBoff0, 0);      // prologue: chunk 0 -> buf 0

  for (int i = 0; i < D_ / 64; ++i) {      // 8 iterations
    const int p = i & 1;
    if (i < D_ / 64 - 1) {
      // issue chunk i+1 into the other buffer, then wait only for chunk i
      stage_tiles(p ? ldsAoff0 : ldsAoff1, p ? ldsBoff0 : ldsBoff1,
                  (i + 1) * 64);
      asm volatile("s_wait_asynccnt 0x6" ::: "memory");
    } else {
      asm volatile("s_wait_asynccnt 0x0" ::: "memory");
    }
    __syncthreads();                       // chunk i visible to all waves

    union FragU { uint4 u[2]; v16bf v; };
#pragma unroll
    for (int h = 0; h < 2; ++h) {          // two K=32 halves of the 64 chunk
      FragU a[2], bfr[2];
#pragma unroll
      for (int tr = 0; tr < 2; ++tr) {
        // A 16x32: lane<16 -> row, K{0-7,16-23}; lane>=16 -> K{8-15,24-31}
        const int r = wm * 32 + tr * 16 + l15;
        a[tr].u[0] = ldsA[p][r * 9 + h * 4 + hi];
        a[tr].u[1] = ldsA[p][r * 9 + h * 4 + 2 + hi];
      }
#pragma unroll
      for (int tc = 0; tc < 2; ++tc) {
        // B 32x16: lane = N column (shot row); low lanes K{0-7,16-23}, high K{8-15,24-31}
        const int r = (wc * 2 + tc) * 16 + l15;
        bfr[tc].u[0] = ldsB[p][r * 9 + h * 4 + hi];
        bfr[tc].u[1] = ldsB[p][r * 9 + h * 4 + 2 + hi];
      }
#pragma unroll
      for (int tr = 0; tr < 2; ++tr)
#pragma unroll
        for (int tc = 0; tc < 2; ++tc)
          acc[tr][tc] = __builtin_amdgcn_wmma_f32_16x16x32_bf16(
              false, a[tr].v, false, bfr[tc].v, (short)0, acc[tr][tc],
              false, false);
    }
    __syncthreads();                       // buf p free for chunk i+2
  }

  // ---- epilogue: score = 2*dot - train_sq[n] - test_sq[m]; max over n ----
  // C layout: VGPR v, lanes 0-15 -> M=v, lanes 16-31 -> M=8+v, N = lane%16.
#pragma unroll
  for (int tr = 0; tr < 2; ++tr) {
#pragma unroll
    for (int tc = 0; tc < 2; ++tc) {
      const int cls = class0 + wc * 2 + tc;
      const float tsq = trainsq[((size_t)b * K_ + cls) * N_ + l15];
#pragma unroll
      for (int v = 0; v < 8; ++v) {
        float val = 2.0f * acc[tr][tc][v] - tsq;
        // max over the 16 shots: butterfly within each 16-lane group
        val = fmaxf(val, __shfl_xor(val, 1));
        val = fmaxf(val, __shfl_xor(val, 2));
        val = fmaxf(val, __shfl_xor(val, 4));
        val = fmaxf(val, __shfl_xor(val, 8));
        if (l15 == v) {   // lanes v and v+16 hold rows v and v+8
          const int m = m0 + wm * 32 + tr * 16 + v + (hi << 3);
          out[((size_t)b * M_ + m) * K_ + cls] =
              val - testsq[(size_t)b * M_ + m];
        }
      }
    }
  }
}

extern "C" void kernel_launch(void* const* d_in, const int* in_sizes, int n_in,
                              void* d_out, int out_size, void* d_ws, size_t ws_size,
                              hipStream_t stream) {
  (void)in_sizes; (void)n_in; (void)out_size; (void)ws_size;
  const float* train = (const float*)d_in[0];   // [B,K,N,D] f32
  const float* test  = (const float*)d_in[1];   // [B,M,D]   f32
  char* ws = (char*)d_ws;
  unsigned short* testbf  = (unsigned short*)(ws + WS_TESTBF);
  unsigned short* trainbf = (unsigned short*)(ws + WS_TRAINBF);
  float* trainsq = (float*)(ws + WS_TRAINSQ);
  float* testsq  = (float*)(ws + WS_TESTSQ);
  float* out = (float*)d_out;

  // 98304 waves / 8 waves-per-block = 12288 blocks
  prep_kernel<<<dim3((B_ * M_ + B_ * K_ * N_) / 8), 256, 0, stream>>>(
      train, test, testbf, trainbf, trainsq, testsq);

  nearest_wmma_kernel<<<dim3(M_ / 128, K_ / 4, B_), 256, 0, stream>>>(
      testbf, trainbf, trainsq, testsq, out);
}
